// DeepseekV3MoE_59691455480111
// MI455X (gfx1250) — matrix operand order repaired and checked by reading, not compiled, not run on hardware
//
#include <hip/hip_runtime.h>
#include <hip/hip_bf16.h>

// ---------------- problem constants ----------------
#define TOK   2048
#define HDIM  768
#define NEXP  128
#define IDIM  1856
#define ISH   3712
#define KTOP  6
#define CAP   192
#define SCALE 2.5f

#define KSTR  40          // LDS row stride in bf16 elems (80B: 16B aligned, conflict-breaking)
#define LDW_GU 3712       // row stride of gate/up weights (2*IDIM == ISH == 3712)
#define LDW_DN 768        // row stride of down weights (== HDIM)

typedef __attribute__((ext_vector_type(16))) __bf16 v16bf;
typedef __attribute__((ext_vector_type(8)))  float  v8f;

struct __align__(32) Pack32 { uint4 lo, hi; };
union  V8U { v8f v; float f[8]; };

// f32 -> bf16 with round-to-nearest bias (1 add + shift)
__device__ __forceinline__ unsigned short f2bf(float f) {
  return (unsigned short)((__builtin_bit_cast(unsigned, f) + 0x8000u) >> 16);
}

// pack two floats into a dword of two bf16 (lo=a, hi=b): 2 adds + 1 v_perm_b32
__device__ __forceinline__ unsigned pk2bf(float a, float b) {
#if defined(__has_builtin) && __has_builtin(__builtin_amdgcn_cvt_pk_bf16_f32)
  typedef __attribute__((ext_vector_type(2))) __bf16 v2bf;
  v2bf r = __builtin_amdgcn_cvt_pk_bf16_f32(a, b);
  return __builtin_bit_cast(unsigned, r);
#else
  unsigned ua = __builtin_bit_cast(unsigned, a) + 0x8000u;
  unsigned ub = __builtin_bit_cast(unsigned, b) + 0x8000u;
  // D = { ub[3], ub[2], ua[3], ua[2] }  (byte sel from {S0=ub, S1=ua})
  return __builtin_amdgcn_perm(ub, ua, 0x07060302u);
#endif
}

__device__ __forceinline__ void wait_async0() {
#if defined(__has_builtin) && __has_builtin(__builtin_amdgcn_s_wait_asynccnt)
  __builtin_amdgcn_s_wait_asynccnt(0);
#else
  asm volatile("s_wait_asynccnt 0x0" ::: "memory");
#endif
}

// per-lane async DMA: 16B global -> LDS, tracked by ASYNCcnt
__device__ __forceinline__ void async_g2l_b128(const void* gptr, void* lptr) {
  unsigned           loff = (unsigned)(uintptr_t)lptr;        // low 32 bits = LDS offset
  unsigned long long ga   = (unsigned long long)(uintptr_t)gptr;
  asm volatile("global_load_async_to_lds_b128 %0, %1, off"
               :: "v"(loff), "v"(ga) : "memory");
}

__device__ __forceinline__ v16bf load_fragA(const unsigned short* p) {
  Pack32 pk;
  pk.lo = *(const uint4*)(p);                  // k .. k+7
  pk.hi = *(const uint4*)(p + 16);             // k+16 .. k+23
  return __builtin_bit_cast(v16bf, pk);
}

__device__ __forceinline__ v16bf load_fragB(const unsigned short* p) {
  Pack32 pk;
  pk.lo = *(const uint4*)(p);                  // k .. k+7
  pk.hi = *(const uint4*)(p + 16);             // k+16 .. k+23
  return __builtin_bit_cast(v16bf, pk);
}

__device__ __forceinline__ v8f wmma_bf16(v16bf a, v16bf b, v8f c) {
  return __builtin_amdgcn_wmma_f32_16x16x32_bf16(false, a, false, b, (short)0, c,
                                                 false, false);
}

// ---------------- 1) router: sigmoid + top-6 + normalize ----------------
__global__ void __launch_bounds__(256)
router_kernel(const float* __restrict__ X, const float* __restrict__ Wg,
              const float* __restrict__ ebias,
              int* __restrict__ topk_idx, float* __restrict__ topk_w)
{
  __shared__ float xs[8][HDIM];
  const int tid  = threadIdx.x;
  const int wv   = tid >> 5;
  const int lane = tid & 31;
  const int t    = blockIdx.x * 8 + wv;

  const float4* xr = (const float4*)(X + (long)t * HDIM);
  for (int k = lane; k < HDIM / 4; k += 32)
    ((float4*)xs[wv])[k] = xr[k];
  __syncthreads();

  float sig[4], choice[4];
  #pragma unroll
  for (int j = 0; j < 4; ++j) {
    const int e = lane * 4 + j;
    const float* wr = Wg + (long)e * HDIM;
    float acc = 0.f;
    for (int k = 0; k < HDIM; k += 4) {
      float4 xv = *(const float4*)&xs[wv][k];
      float4 wvv = *(const float4*)&wr[k];
      acc += xv.x * wvv.x + xv.y * wvv.y + xv.z * wvv.z + xv.w * wvv.w;
    }
    float s   = 1.f / (1.f + __expf(-acc));
    sig[j]    = s;
    choice[j] = s + ebias[e];
  }

  bool taken[4] = {false, false, false, false};
  int   widx[KTOP];
  float wsg[KTOP];
  float sum = 0.f;
  for (int it = 0; it < KTOP; ++it) {
    float m = -1e30f, msv = 0.f; int mi = -1;
    #pragma unroll
    for (int j = 0; j < 4; ++j)
      if (!taken[j] && choice[j] > m) { m = choice[j]; msv = sig[j]; mi = lane * 4 + j; }
    for (int off = 16; off > 0; off >>= 1) {
      float om  = __shfl_xor(m, off);
      float osv = __shfl_xor(msv, off);
      int   omi = __shfl_xor(mi, off);
      if (om > m || (om == m && omi >= 0 && (mi < 0 || omi < mi))) { m = om; msv = osv; mi = omi; }
    }
    widx[it] = mi; wsg[it] = msv; sum += msv;
    if (mi >= 0 && (mi >> 2) == lane) taken[mi & 3] = true;
  }

  if (lane == 0) {
    const float inv = SCALE / (sum + 1e-20f);
    #pragma unroll
    for (int it = 0; it < KTOP; ++it) {
      topk_idx[t * KTOP + it] = widx[it];
      topk_w  [t * KTOP + it] = wsg[it] * inv;
    }
  }
}

// ---------------- 2) capacity dispatch ----------------
__global__ void __launch_bounds__(256)
dispatch_kernel(const int* __restrict__ topk_idx, const float* __restrict__ topk_w,
                int* __restrict__ counts, int* __restrict__ token_buf,
                float* __restrict__ w_buf)
{
  const int p = blockIdx.x * 256 + threadIdx.x;
  if (p >= TOK * KTOP) return;
  const int   e = topk_idx[p];
  const float w = topk_w[p];
  const int pos = atomicAdd(&counts[e], 1);
  if (pos < CAP) {
    token_buf[e * CAP + pos] = p / KTOP;
    w_buf    [e * CAP + pos] = w;
  }
}

// ---------------- 3) fused gate+up GEMM (bf16 WMMA), silu*u epilogue ----------------
// Out[row, col] = silu(Xg @ Wg) * (Xg @ Wu), 64x64 tile per block, K = HDIM fixed,
// weight row stride fixed (LDW_GU) so row offsets fold into instruction offsets.
__global__ void __launch_bounds__(256)
gemm_gate_up(const float* __restrict__ X,
             const int* __restrict__ gather, long gatherz,
             const float* __restrict__ Wgp, const float* __restrict__ Wup,
             long wz,
             unsigned short* __restrict__ Out, long ldo, long outz)
{
  __shared__ __align__(16) unsigned short As[64 * KSTR];
  __shared__ __align__(16) unsigned short Bg[64 * KSTR];
  __shared__ __align__(16) unsigned short Bu[64 * KSTR];

  const int tid = threadIdx.x;
  const int z   = blockIdx.z;
  const int rowBase = blockIdx.y * 64;
  const int colBase = blockIdx.x * 64;

  const float* wg = Wgp + (long)z * wz;
  const float* wu = Wup + (long)z * wz;
  unsigned short* out = Out + (long)z * outz;

  // A staging: thread -> (row, 8 contiguous k)
  const int arow = tid >> 2;
  const int ak   = (tid & 3) * 8;
  const int atok = gather ? gather[(long)z * gatherz + rowBase + arow]
                          : (rowBase + arow);

  // B staging: thread -> (4 contiguous k) x (2 contiguous n); packed b64 LDS stores
  const int bn = (tid & 31) * 2;     // n, n+1
  const int bk = (tid >> 5) * 4;     // k .. k+3

  const int wv = tid >> 5, lane = tid & 31;
  const int half = lane >> 4, r16 = lane & 15;
  const int n0 = (wv & 3) * 16;
  const int m0 = (wv >> 2) * 16;
  const int akb = half * 8;
  const int bkb = half * 8;

  v8f accG0 = {0,0,0,0,0,0,0,0}, accG1 = {0,0,0,0,0,0,0,0};
  v8f accU0 = {0,0,0,0,0,0,0,0}, accU1 = {0,0,0,0,0,0,0,0};

  // pipeline registers + running pointers (constant strides fold into offsets)
  float4 ra0, ra1;
  float2 rg[4], ru[4];
  const float* arp = X + (long)atok * HDIM + ak;
  const float* grp = wg + (long)bk * LDW_GU + colBase + bn;
  const float* urp = wu + (long)bk * LDW_GU + colBase + bn;

  auto preload = [&](int kk) {
    ra0 = *(const float4*)(arp);
    ra1 = *(const float4*)(arp + 4);
    #pragma unroll
    for (int r = 0; r < 4; ++r) {
      rg[r] = *(const float2*)(grp + r * LDW_GU);
      ru[r] = *(const float2*)(urp + r * LDW_GU);
    }
    if (kk + 128 < HDIM) {           // L2 streaming hint, 4 tiles ahead
      __builtin_prefetch(grp + 128 * LDW_GU, 0, 1);
      __builtin_prefetch(urp + 128 * LDW_GU, 0, 1);
    }
    arp += 32; grp += 32 * LDW_GU; urp += 32 * LDW_GU;
  };

  preload(0);

  for (int kk = 0; kk < HDIM; kk += 32) {
    // ---- stage A (fp32 -> bf16, one b128 store) ----
    uint4 ap;
    ap.x = pk2bf(ra0.x, ra0.y); ap.y = pk2bf(ra0.z, ra0.w);
    ap.z = pk2bf(ra1.x, ra1.y); ap.w = pk2bf(ra1.z, ra1.w);
    *(uint4*)&As[arow * KSTR + ak] = ap;
    // ---- stage B gate/up transposed (two b64 stores each) ----
    uint2 v;
    v.x = pk2bf(rg[0].x, rg[1].x); v.y = pk2bf(rg[2].x, rg[3].x);
    *(uint2*)&Bg[(bn + 0) * KSTR + bk] = v;
    v.x = pk2bf(rg[0].y, rg[1].y); v.y = pk2bf(rg[2].y, rg[3].y);
    *(uint2*)&Bg[(bn + 1) * KSTR + bk] = v;
    v.x = pk2bf(ru[0].x, ru[1].x); v.y = pk2bf(ru[2].x, ru[3].x);
    *(uint2*)&Bu[(bn + 0) * KSTR + bk] = v;
    v.x = pk2bf(ru[0].y, ru[1].y); v.y = pk2bf(ru[2].y, ru[3].y);
    *(uint2*)&Bu[(bn + 1) * KSTR + bk] = v;
    __syncthreads();

    if (kk + 32 < HDIM) preload(kk + 32);   // overlap HBM with WMMA

    v16bf af0 = load_fragA(&As[(m0      + r16) * KSTR + akb]);
    v16bf af1 = load_fragA(&As[(m0 + 32 + r16) * KSTR + akb]);
    v16bf bg  = load_fragB(&Bg[(n0 + r16) * KSTR + bkb]);
    v16bf bu  = load_fragB(&Bu[(n0 + r16) * KSTR + bkb]);

    accG0 = wmma_bf16(af0, bg, accG0);
    accG1 = wmma_bf16(af1, bg, accG1);
    accU0 = wmma_bf16(af0, bu, accU0);
    accU1 = wmma_bf16(af1, bu, accU1);
    __syncthreads();
  }

  V8U G[2], U[2];
  G[0].v = accG0; G[1].v = accG1; U[0].v = accU0; U[1].v = accU1;
  const int col = colBase + n0 + r16;
  #pragma unroll
  for (int s = 0; s < 2; ++s) {
    const int mrow = rowBase + m0 + s * 32 + half * 8;
    #pragma unroll
    for (int r = 0; r < 8; ++r) {
      float g = G[s].f[r], u = U[s].f[r];
      float sv = g / (1.f + __expf(-g)) * u;       // silu(g) * u
      out[(long)(mrow + r) * ldo + col] = f2bf(sv);
    }
  }
}

// ---------------- 4) down-proj GEMM (bf16 WMMA) + weighted scatter-add ----------------
// A (already bf16) is DMA'd global->LDS with async-to-LDS, double buffered;
// B (fp32 weights, row stride fixed at HDIM) converted through registers, pipelined.
__global__ void __launch_bounds__(256)
gemm_down(const unsigned short* __restrict__ A, long lda, long az,
          const float* __restrict__ W, long wz,
          const int* __restrict__ scatter, long scatz,
          const float* __restrict__ wscale,
          float* __restrict__ Out, int Kd)
{
  __shared__ __align__(16) unsigned short As[2][64 * KSTR];
  __shared__ __align__(16) unsigned short Bs[64 * KSTR];

  const int tid = threadIdx.x;
  const int z   = blockIdx.z;
  const int rowBase = blockIdx.y * 64;
  const int colBase = blockIdx.x * 64;

  const unsigned short* a = A + (long)z * az;
  const int*   sct = scatter ? scatter + (long)z * scatz : nullptr;
  const float* wsc = wscale  ? wscale  + (long)z * scatz : nullptr;

  const int arow = tid >> 2;
  const int ak   = (tid & 3) * 8;
  const int bn   = (tid & 31) * 2;
  const int bk   = (tid >> 5) * 4;

  const int wv = tid >> 5, lane = tid & 31;
  const int half = lane >> 4, r16 = lane & 15;
  const int n0 = (wv & 3) * 16;
  const int m0 = (wv >> 2) * 16;
  const int akb = half * 8;
  const int bkb = half * 8;

  v8f acc0 = {0,0,0,0,0,0,0,0}, acc1 = {0,0,0,0,0,0,0,0};

  const unsigned short* arp = a + (long)(rowBase + arow) * lda + ak;
  const float*          brp = W + (long)z * wz + (long)bk * LDW_DN + colBase + bn;

  float2 rb[4];
  auto preloadB = [&](int kk) {
    #pragma unroll
    for (int r = 0; r < 4; ++r)
      rb[r] = *(const float2*)(brp + r * LDW_DN);
    if (kk + 128 < Kd) __builtin_prefetch(brp + 128 * LDW_DN, 0, 1);
    brp += 32 * LDW_DN;
  };
  auto issueA = [&](int buf) {       // 16B per lane, ASYNCcnt-tracked
    async_g2l_b128(arp, &As[buf][arow * KSTR + ak]);
    arp += 32;
  };

  issueA(0);
  preloadB(0);

  for (int kk = 0; kk < Kd; kk += 32) {
    const int buf = (kk >> 5) & 1;
    // ---- stage B (fp32 -> bf16, transposed, packed b64 stores) ----
    uint2 v;
    v.x = pk2bf(rb[0].x, rb[1].x); v.y = pk2bf(rb[2].x, rb[3].x);
    *(uint2*)&Bs[(bn + 0) * KSTR + bk] = v;
    v.x = pk2bf(rb[0].y, rb[1].y); v.y = pk2bf(rb[2].y, rb[3].y);
    *(uint2*)&Bs[(bn + 1) * KSTR + bk] = v;

    wait_async0();                 // A tile for this kk has landed in LDS
    __syncthreads();

    if (kk + 32 < Kd) {            // overlap next A DMA + B fetch with WMMA
      issueA(buf ^ 1);
      preloadB(kk + 32);
    }

    v16bf af0 = load_fragA(&As[buf][(m0      + r16) * KSTR + akb]);
    v16bf af1 = load_fragA(&As[buf][(m0 + 32 + r16) * KSTR + akb]);
    v16bf bf  = load_fragB(&Bs[(n0 + r16) * KSTR + bkb]);

    acc0 = wmma_bf16(af0, bf, acc0);
    acc1 = wmma_bf16(af1, bf, acc1);
    __syncthreads();
  }

  V8U Y[2];
  Y[0].v = acc0; Y[1].v = acc1;
  const int col = colBase + n0 + r16;
  #pragma unroll
  for (int s = 0; s < 2; ++s) {
    const int mrow = rowBase + m0 + s * 32 + half * 8;
    #pragma unroll
    for (int r = 0; r < 8; ++r) {
      const int row = mrow + r;
      const int tok = sct ? sct[row] : row;
      const float sc = wsc ? wsc[row] : 1.0f;
      atomicAdd(&Out[(long)tok * HDIM + col], Y[s].f[r] * sc);
    }
  }
}

// ---------------- host orchestration ----------------
extern "C" void kernel_launch(void* const* d_in, const int* in_sizes, int n_in,
                              void* d_out, int out_size, void* d_ws, size_t ws_size,
                              hipStream_t stream) {
  (void)in_sizes; (void)n_in; (void)out_size; (void)ws_size;
  const float* X    = (const float*)d_in[0];   // [1,2048,768]
  const float* Wg   = (const float*)d_in[1];   // [128,768]
  const float* eb   = (const float*)d_in[2];   // [128]
  const float* Wgu  = (const float*)d_in[3];   // [128,768,3712]
  const float* Wd   = (const float*)d_in[4];   // [128,1856,768]
  const float* Wsg  = (const float*)d_in[5];   // [768,3712]
  const float* Wsu  = (const float*)d_in[6];   // [768,3712]
  const float* Wsd  = (const float*)d_in[7];   // [3712,768]
  float* out = (float*)d_out;

  char* ws = (char*)d_ws;
  int*            topk_idx  = (int*)  (ws + 0);                    // 12288 * 4
  float*          topk_w    = (float*)(ws + 49152);                // 12288 * 4
  int*            counts    = (int*)  (ws + 98304);                // 128 * 4
  int*            token_buf = (int*)  (ws + 98816);                // 24576 * 4
  float*          w_buf     = (float*)(ws + 197120);               // 24576 * 4
  unsigned short* inter     = (unsigned short*)(ws + 295424);      // 128*192*1856 bf16
  unsigned short* s_inter   = (unsigned short*)(ws + 295424 + 91226112); // 2048*3712 bf16

  hipMemsetAsync(counts,    0, 128 * sizeof(int), stream);
  hipMemsetAsync(token_buf, 0, NEXP * CAP * sizeof(int), stream);
  hipMemsetAsync(w_buf,     0, NEXP * CAP * sizeof(float), stream);
  hipMemsetAsync(out,       0, (size_t)TOK * HDIM * sizeof(float), stream);

  router_kernel<<<TOK / 8, 256, 0, stream>>>(X, Wg, eb, topk_idx, topk_w);
  dispatch_kernel<<<(TOK * KTOP) / 256, 256, 0, stream>>>(topk_idx, topk_w,
                                                          counts, token_buf, w_buf);

  // routed experts: gate_up  (cols 29x64=1856, rows 3x64=192, z=128 experts)
  gemm_gate_up<<<dim3(IDIM / 64, CAP / 64, NEXP), 256, 0, stream>>>(
      X, token_buf, (long)CAP,
      Wgu, Wgu + IDIM, (long)HDIM * 2 * IDIM,
      inter, (long)IDIM, (long)CAP * IDIM);

  // routed experts: down + weighted scatter-add
  gemm_down<<<dim3(HDIM / 64, CAP / 64, NEXP), 256, 0, stream>>>(
      inter, (long)IDIM, (long)CAP * IDIM,
      Wd, (long)IDIM * HDIM,
      token_buf, (long)CAP, w_buf, out, IDIM);

  // shared expert: gate_up  (cols 58x64=3712, rows 32x64=2048)
  gemm_gate_up<<<dim3(ISH / 64, TOK / 64, 1), 256, 0, stream>>>(
      X, nullptr, 0,
      Wsg, Wsu, 0,
      s_inter, (long)ISH, 0);

  // shared expert: down + add
  gemm_down<<<dim3(HDIM / 64, TOK / 64, 1), 256, 0, stream>>>(
      s_inter, (long)ISH, 0,
      Wsd, 0,
      nullptr, 0, nullptr, out, ISH);
}
